// ClusteringLoss_38319698215236
// MI455X (gfx1250) — compile-verified, matrix operand order checked by reading, not run on hardware
//
#include <hip/hip_runtime.h>
#include <hip/hip_bf16.h>

typedef __attribute__((ext_vector_type(16))) _Float16 v16h;
typedef __attribute__((ext_vector_type(8)))  float    v8f;

#define CLS 32
#define EMB 16

// ws float-offset map:
//   0: counts[32]      32: sumf[32*16]    544: sumv[32]
// 576: var_c[32]      608: inv2v[32]      640: m2[32]
// 672: sm_num[32]     704: bce_sum        705: seed_sum   706: const_part
// 768: bl0 (512 f16 = cluster-tile-0 operand)   1024: bl1 (512 f16)

__global__ void k_zero(float* ws) {
    int i = blockIdx.x * blockDim.x + threadIdx.x;
    if (i < 768) ws[i] = 0.0f;
}

// Pass 1: segmented sums (counts, sum features, sum exp(s1)) via LDS atomics.
__global__ __launch_bounds__(256) void k_seg(const float* __restrict__ feat,
                                             const float* __restrict__ seed,
                                             const int* __restrict__ lab,
                                             float* __restrict__ ws, int N) {
    __shared__ float sc[CLS];
    __shared__ float sv[CLS];
    __shared__ float sf[CLS * EMB];
    for (int t = threadIdx.x; t < CLS * EMB; t += blockDim.x) sf[t] = 0.0f;
    if (threadIdx.x < CLS) { sc[threadIdx.x] = 0.0f; sv[threadIdx.x] = 0.0f; }
    __syncthreads();

    int stride = gridDim.x * blockDim.x;
    for (int p = blockIdx.x * blockDim.x + threadIdx.x; p < N; p += stride) {
        int c = lab[p];
        atomicAdd(&sc[c], 1.0f);
        atomicAdd(&sv[c], __expf(seed[2 * (size_t)p + 1]));
        const float4* f4 = (const float4*)(feat + (size_t)p * EMB);
        float4 x0 = f4[0], x1 = f4[1], x2 = f4[2], x3 = f4[3];
        float* dst = &sf[c * EMB];
        atomicAdd(dst + 0,  x0.x); atomicAdd(dst + 1,  x0.y);
        atomicAdd(dst + 2,  x0.z); atomicAdd(dst + 3,  x0.w);
        atomicAdd(dst + 4,  x1.x); atomicAdd(dst + 5,  x1.y);
        atomicAdd(dst + 6,  x1.z); atomicAdd(dst + 7,  x1.w);
        atomicAdd(dst + 8,  x2.x); atomicAdd(dst + 9,  x2.y);
        atomicAdd(dst + 10, x2.z); atomicAdd(dst + 11, x2.w);
        atomicAdd(dst + 12, x3.x); atomicAdd(dst + 13, x3.y);
        atomicAdd(dst + 14, x3.z); atomicAdd(dst + 15, x3.w);
    }
    __syncthreads();
    for (int t = threadIdx.x; t < CLS * EMB; t += blockDim.x)
        atomicAdd(&ws[32 + t], sf[t]);
    if (threadIdx.x < CLS) {
        atomicAdd(&ws[threadIdx.x], sc[threadIdx.x]);
        atomicAdd(&ws[544 + threadIdx.x], sv[threadIdx.x]);
    }
}

// Phase between passes: means, var, m2, 1/(2var), centroid-pair dist loss, reg
// loss, and pack cmeans into the WMMA f16 per-lane operand layout.
// Note: the 16-bit A layout (lane M: K0..7, lane M+16: K8..15, K16..31 = 0)
// and B layout (lane N: K0..7, lane N+16: K8..15) are the same per-lane byte
// pattern, so this packing serves as the A operand of cmeans @ features^T.
__global__ void k_means(float* ws) {
    __shared__ float mns[CLS][EMB];
    int c = threadIdx.x;   // 32 threads
    float inv = 1.0f / ws[c];
    float m2s = 0.0f;
    for (int k = 0; k < EMB; ++k) {
        float mk = ws[32 + c * EMB + k] * inv;
        mns[c][k] = mk;
        m2s += mk * mk;
    }
    float var = ws[544 + c] * inv;
    ws[576 + c] = var;
    ws[608 + c] = 0.5f / var;   // 1/(2*var)
    ws[640 + c] = m2s;
    __syncthreads();

    // inter-cluster hinge loss over off-diagonal pairs
    float dacc = 0.0f;
    for (int b = 0; b < CLS; ++b) {
        if (b == c) continue;
        float d2 = 0.0f;
        for (int k = 0; k < EMB; ++k) {
            float df = mns[c][k] - mns[b][k];
            d2 += df * df;
        }
        float h = fmaxf(2.0f * 1.5f - sqrtf(d2), 0.0f);
        dacc += h * h;
    }
    float reg = sqrtf(m2s);
    for (int o = 16; o > 0; o >>= 1) {
        dacc += __shfl_xor(dacc, o, 32);
        reg  += __shfl_xor(reg,  o, 32);
    }
    if (c == 0)
        ws[706] = 1.0f * dacc / (31.0f * 32.0f) + 0.001f * reg / 32.0f;

    _Float16* bl = (_Float16*)(ws + ((c < 16) ? 768 : 1024));
    int n = c & 15;
    for (int k = 0; k < 8; ++k) {
        bl[n * 16 + k]            = (_Float16)mns[c][k];
        bl[n * 16 + 8 + k]        = (_Float16)0.0f;
        bl[(n + 16) * 16 + k]     = (_Float16)mns[c][8 + k];
        bl[(n + 16) * 16 + 8 + k] = (_Float16)0.0f;
    }
}

// Pass 2: one wave per 16-point tile. D = cmeans @ features^T, so D rows are
// clusters and D columns are points: lane (m, half) holds clusters
// {8*half + r} x its OWN point base+m -> epilogue is fully lane-local
// (no shuffles in the unrolled r-loop).
__global__ __launch_bounds__(256) void k_main(const float* __restrict__ feat,
                                              const float* __restrict__ seed,
                                              const int* __restrict__ lab,
                                              float* ws, int ntiles) {
    __shared__ float smLds[CLS];
    __shared__ float varLds[CLS];
    if (threadIdx.x < CLS) {
        smLds[threadIdx.x] = 0.0f;
        varLds[threadIdx.x] = ws[576 + threadIdx.x];
    }
    __syncthreads();

    const int lane = threadIdx.x & 31;
    const int half = lane >> 4;      // 0: emb 0..7 / clusters 0..7(+16), 1: emb 8..15 / clusters 8..15(+16)
    const int m    = lane & 15;      // point-in-tile (D column)
    const int wid  = blockIdx.x * (blockDim.x >> 5) + (threadIdx.x >> 5);
    const int nw   = gridDim.x * (blockDim.x >> 5);

    // A operands (constant across tiles): cluster tiles 0-15 and 16-31.
    const v16h a0 = *(const v16h*)((const _Float16*)(ws + 768)  + lane * 16);
    const v16h a1 = *(const v16h*)((const _Float16*)(ws + 1024) + lane * 16);

    // Per-cluster constants for this lane's D rows: cluster = 8*half + r (+16).
    const int cbase = 8 * half;
    float m2_t0[8], m2_t1[8], i2_t0[8], i2_t1[8];
    #pragma unroll
    for (int r = 0; r < 8; ++r) {
        m2_t0[r] = ws[640 + cbase + r];
        m2_t1[r] = ws[640 + 16 + cbase + r];
        i2_t0[r] = ws[608 + cbase + r];
        i2_t1[r] = ws[608 + 16 + cbase + r];
    }

    float bce_acc = 0.0f, seed_acc = 0.0f;

    for (int t = wid; t < ntiles; t += nw) {
        const int base = t * 16;
        const int p = base + m;
        // B operand (features^T, K x Npoints): lane m -> feat[p][0..7] in
        // halves 0..7; lane m+16 -> feat[p][8..15]; K 16..31 zero.
        const float* fp = feat + (size_t)p * EMB + half * 8;
        float4 x0 = ((const float4*)fp)[0];
        float4 x1 = ((const float4*)fp)[1];
        float f2p = x0.x * x0.x + x0.y * x0.y + x0.z * x0.z + x0.w * x0.w
                  + x1.x * x1.x + x1.y * x1.y + x1.z * x1.z + x1.w * x1.w;
        float f2m = f2p + __shfl_xor(f2p, 16, 32);   // full |feat[p]|^2

        v16h b = {};
        b[0] = (_Float16)x0.x; b[1] = (_Float16)x0.y;
        b[2] = (_Float16)x0.z; b[3] = (_Float16)x0.w;
        b[4] = (_Float16)x1.x; b[5] = (_Float16)x1.y;
        b[6] = (_Float16)x1.z; b[7] = (_Float16)x1.w;

        int   lb = lab[p];
        float2 sd = *(const float2*)(seed + 2 * (size_t)p);
        float s0 = sd.x;
        float vv = __expf(sd.y);

        v8f c0 = {}, c1 = {};
        c0 = __builtin_amdgcn_wmma_f32_16x16x32_f16(false, a0, false, b,
                                                    (short)0, c0, false, false);
        c1 = __builtin_amdgcn_wmma_f32_16x16x32_f16(false, a1, false, b,
                                                    (short)0, c1, false, false);

        // smoothness: one contribution per point (half 0 only)
        if (half == 0) {
            float dv = vv - varLds[lb];
            atomicAdd(&smLds[lb], dv * dv);
        }

        // D layout: vgpr r -> row (cluster) cbase + r, column (point) = m.
        // Everything is lane-local; exactly one (r, tile) matches the label.
        #pragma unroll
        for (int r = 0; r < 8; ++r) {
            const int cc = cbase + r;
            {   // cluster tile 0: cluster = cc
                float G   = c0[r];
                float D2  = fmaxf(f2m + m2_t0[r] - 2.0f * G, 0.0f);
                float lp  = -D2 * i2_t0[r];
                float P   = __expf(lp);
                bool  y   = (lb == cc);
                float lpc = fmaxf(lp, -100.0f);
                float lmp = fmaxf(__logf(1.0f - P), -100.0f);
                float e   = P - s0;
                bce_acc  -= y ? lpc : lmp;
                seed_acc += y ? e * e : 0.0f;
            }
            {   // cluster tile 1: cluster = cc + 16
                float G   = c1[r];
                float D2  = fmaxf(f2m + m2_t1[r] - 2.0f * G, 0.0f);
                float lp  = -D2 * i2_t1[r];
                float P   = __expf(lp);
                bool  y   = (lb == cc + 16);
                float lpc = fmaxf(lp, -100.0f);
                float lmp = fmaxf(__logf(1.0f - P), -100.0f);
                float e   = P - s0;
                bce_acc  -= y ? lpc : lmp;
                seed_acc += y ? e * e : 0.0f;
            }
        }
    }

    for (int o = 16; o > 0; o >>= 1) {
        bce_acc  += __shfl_xor(bce_acc,  o, 32);
        seed_acc += __shfl_xor(seed_acc, o, 32);
    }
    if (lane == 0) {
        atomicAdd(&ws[704], bce_acc);
        atomicAdd(&ws[705], seed_acc);
    }
    __syncthreads();
    if (threadIdx.x < CLS)
        atomicAdd(&ws[672 + threadIdx.x], smLds[threadIdx.x]);
}

__global__ void k_final(const float* ws, float* out, int N) {
    int c = threadIdx.x;   // 32 threads
    float smc = ws[672 + c] / ws[c];
    for (int o = 16; o > 0; o >>= 1) smc += __shfl_xor(smc, o, 32);
    if (c == 0) {
        float loss = 3.0f * (ws[704] / ((float)N * 32.0f))   // W_VAR * var_loss
                   + ws[706]                                  // dist + reg
                   + 5.0f * (smc / 32.0f)                     // W_SMOOTH
                   + 5.0f * (ws[705] / (float)N);             // W_SEED
        out[0] = loss;
    }
}

extern "C" void kernel_launch(void* const* d_in, const int* in_sizes, int n_in,
                              void* d_out, int out_size, void* d_ws, size_t ws_size,
                              hipStream_t stream) {
    const float* feat = (const float*)d_in[0];
    const float* seed = (const float*)d_in[1];
    const int*   lab  = (const int*)d_in[2];
    float* ws  = (float*)d_ws;
    float* out = (float*)d_out;
    int N = in_sizes[0] / EMB;          // 1,000,000 (divisible by 16)
    int ntiles = N / 16;

    k_zero <<<3, 256, 0, stream>>>(ws);
    k_seg  <<<1024, 256, 0, stream>>>(feat, seed, lab, ws, N);
    k_means<<<1, 32, 0, stream>>>(ws);
    k_main <<<1024, 256, 0, stream>>>(feat, seed, lab, ws, ntiles);
    k_final<<<1, 32, 0, stream>>>(ws, out, N);
}
